// CorrespodenceNet_66889820668453
// MI455X (gfx1250) — matrix operand order repaired and checked by reading, not compiled, not run on hardware
//
#include <hip/hip_runtime.h>
#include <hip/hip_bf16.h>

// CDNA5 / gfx1250, wave32. Fused correlation-softmax-warp ("CorrespondenceNet").
//
//   xf = colnorm(x)  [N,C] -> f16        (phase 1, workspace)
//   yf = colnorm(y)  [N,C] -> f16
//   flash loop over column tiles:  corr tile via V_WMMA_F32_16X16X32_F16,
//   per-LANE online softmax state (no per-tile cross-lane reductions);
//   one cross-lane flash combine at the end. The N x N correlation matrix
//   (1 GB) is never materialized. B operand is software-pipelined 4 chunks
//   deep so global_load_b128 latency overlaps WMMA execution.
//
// N = 16384, C = 256 for this problem instance (C must be 256: 8 K-steps).

typedef __attribute__((ext_vector_type(16))) _Float16 v16h;
typedef __attribute__((ext_vector_type(8)))  _Float16 h8;
typedef __attribute__((ext_vector_type(8)))  float    v8f;

#define K2 144.26950408889634f   // (1/tau) * log2(e) = 100 * 1.4426950408889634

// ---------------------------------------------------------------------------
// Phase 1: per-channel mean-center + L2 normalize (over the N positions),
// emit f16. One block per (channel, which-tensor).
// ---------------------------------------------------------------------------
__global__ void normalize_cols(const float* __restrict__ x,
                               const float* __restrict__ y,
                               _Float16* __restrict__ xh,
                               _Float16* __restrict__ yh,
                               int N, int C)
{
    const float* f = (blockIdx.y == 0) ? x : y;
    _Float16*    o = (blockIdx.y == 0) ? xh : yh;
    const int c   = blockIdx.x;
    const int tid = threadIdx.x;
    __shared__ float red[256];

    float s = 0.f;
    for (int i = tid; i < N; i += 256) s += f[(size_t)i * C + c];
    red[tid] = s; __syncthreads();
    for (int off = 128; off > 0; off >>= 1) {
        if (tid < off) red[tid] += red[tid + off];
        __syncthreads();
    }
    const float mean = red[0] / (float)N;
    __syncthreads();

    float s2 = 0.f;
    for (int i = tid; i < N; i += 256) {
        float d = f[(size_t)i * C + c] - mean;
        s2 += d * d;
    }
    red[tid] = s2; __syncthreads();
    for (int off = 128; off > 0; off >>= 1) {
        if (tid < off) red[tid] += red[tid + off];
        __syncthreads();
    }
    const float inv = 1.0f / sqrtf(red[0]);

    for (int i = tid; i < N; i += 256) {
        float d = (f[(size_t)i * C + c] - mean) * inv;
        o[(size_t)i * C + c] = (_Float16)d;
    }
}

// ---------------------------------------------------------------------------
// Phase 2: fused corr / softmax / warp. One wave per 16-row tile; A panel
// (16 x 256 f16, 64 VGPRs) resident; sweep 1024 column tiles of B with a
// 4-chunk-deep B register pipeline (Bq, 32 VGPRs).
//
// WMMA f16 operand layouts (CDNA5 ISA, wave32):
//   A 16x32:  lane L -> M = L&15; lanes<16 hold K {0..7,16..23},
//             lanes>=16 hold K {8..15,24..31}; packed consecutive-K pairs.
//   B 32x16:  lane L -> Ncol = L&15; same K order as A. This code loads
//             both A and B as lanes<16 K 0..15, lanes>=16 K 16..31 (contiguous).
//   C/D f32:  VGPR r, lane<16 -> (M=r,   N=lane),
//                     lane>=16 -> (M=r+8, N=lane-16).
// Both operands = two global_load_b128 per lane per K-step (row-major, no
// transpose needed since B = yf^T and contraction is over C).
// ---------------------------------------------------------------------------
__global__ void __launch_bounds__(256)
corr_softmax_warp(const _Float16* __restrict__ xh,
                  const _Float16* __restrict__ yh,
                  const float* __restrict__ ref_ab,
                  float* __restrict__ out,
                  int N, int C)
{
    const int lane    = threadIdx.x & 31;
    const int wave    = threadIdx.x >> 5;
    const int rowTile = blockIdx.x * (blockDim.x >> 5) + wave;
    const int rowBase = rowTile * 16;
    const int nLocal  = lane & 15;
    const int hiHalf  = lane >> 4;       // 0 or 1

    // ---- load resident A panel: rows rowBase..rowBase+15, K = 0..255 ----
    v16h A[8];
    {
        const _Float16* pA = xh + (size_t)(rowBase + nLocal) * C + hiHalf * 16;
        #pragma unroll
        for (int ks = 0; ks < 8; ++ks) {
            h8 lo = *(const h8*)(pA + ks * 32);        // K = 32ks + 16*hi + 0..7
            h8 hi = *(const h8*)(pA + ks * 32 + 8);    // K = 32ks + 16*hi + 8..15
            v16h a;
            #pragma unroll
            for (int e = 0; e < 8; ++e) { a[e] = lo[e]; a[e + 8] = hi[e]; }
            A[ks] = a;
        }
    }

    // per-LANE flash-softmax state (each lane covers columns == nLocal mod 16)
    float m[8], sum[8], accA[8], accB[8];
    #pragma unroll
    for (int r = 0; r < 8; ++r) {
        m[r] = -1e30f; sum[r] = 0.f; accA[r] = 0.f; accB[r] = 0.f;
    }

    const float* refA = ref_ab;          // ref_ab[0, :, :] flattened
    const float* refB = ref_ab + N;      // ref_ab[1, :, :]

    const int numColTiles = N / 16;
    const _Float16* pBlane = yh + (size_t)nLocal * C + hiHalf * 16;

    // one B chunk = one 16x32 WMMA operand for this lane (two b128 loads)
    auto loadChunk = [&](const _Float16* p) -> v16h {
        h8 lo = *(const h8*)(p);         // K block +0..7
        h8 hi = *(const h8*)(p + 8);     // K block +8..15
        v16h b;
        #pragma unroll
        for (int e = 0; e < 8; ++e) { b[e] = lo[e]; b[e + 8] = hi[e]; }
        return b;
    };

    // prime the 4-deep B pipeline with tile 0, chunks 0..3
    v16h Bq[4];
    #pragma unroll
    for (int k = 0; k < 4; ++k) Bq[k] = loadChunk(pBlane + k * 32);

    for (int j = 0; j < numColTiles; ++j) {
        const _Float16* pBj = pBlane + (size_t)j * 16 * C;
        const int jn = (j + 1 < numColTiles) ? (j + 1) : j;   // clamp, no branch
        const _Float16* pBjn = pBlane + (size_t)jn * 16 * C;

        if (j + 2 < numColTiles)                              // global_prefetch_b8
            __builtin_prefetch(pBlane + (size_t)(j + 2) * 16 * C, 0, 1);

        // ---- 16x16 corr tile, K = 256 via 8 WMMAs; refill 4 slots ahead ----
        v8f acc = {};
        #pragma unroll
        for (int ks = 0; ks < 8; ++ks) {
            v16h bcur = Bq[ks & 3];
            Bq[ks & 3] = (ks < 4) ? loadChunk(pBj  + (ks + 4) * 32)   // (j,   ks+4)
                                  : loadChunk(pBjn + (ks - 4) * 32);  // (j+1, ks-4)
            acc = __builtin_amdgcn_wmma_f32_16x16x32_f16(
                      false, A[ks], false, bcur, (short)0, acc, false, false);
        }

        // ---- per-lane online softmax update: no cross-lane ops ----
        const float ra = refA[j * 16 + nLocal];
        const float rb = refB[j * 16 + nLocal];

        #pragma unroll
        for (int r = 0; r < 8; ++r) {
            const float v     = acc[r];
            const float mOld  = m[r];
            const float mNew  = fmaxf(mOld, v);
            const float scale = exp2f((mOld - mNew) * K2);   // v_exp_f32 (TRANS)
            const float p     = exp2f((v    - mNew) * K2);   // v_exp_f32 (TRANS)
            sum[r]  = sum[r]  * scale + p;
            accA[r] = accA[r] * scale + p * ra;
            accB[r] = accB[r] * scale + p * rb;
            m[r]    = mNew;
        }
    }

    // ---- one-time flash combine across the 16-lane half-wave ----
    #pragma unroll
    for (int r = 0; r < 8; ++r) {
        float M = m[r];
        M = fmaxf(M, __shfl_xor(M, 1));
        M = fmaxf(M, __shfl_xor(M, 2));
        M = fmaxf(M, __shfl_xor(M, 4));
        M = fmaxf(M, __shfl_xor(M, 8));
        const float f = exp2f((m[r] - M) * K2);   // rescale lane partials to M
        float s = sum[r] * f, a = accA[r] * f, b = accB[r] * f;
        s += __shfl_xor(s, 1); a += __shfl_xor(a, 1); b += __shfl_xor(b, 1);
        s += __shfl_xor(s, 2); a += __shfl_xor(a, 2); b += __shfl_xor(b, 2);
        s += __shfl_xor(s, 4); a += __shfl_xor(a, 4); b += __shfl_xor(b, 4);
        s += __shfl_xor(s, 8); a += __shfl_xor(a, 8); b += __shfl_xor(b, 8);

        // lane 0 owns row r, lane 16 owns row r+8
        if (nLocal == 0) {
            const int row = rowBase + r + hiHalf * 8;
            const float is = 1.0f / s;
            out[row]         = a * is;   // W_out a-channel
            out[N + row]     = b * is;   // W_out b-channel
            out[2 * N + row] = M;        // confidence S_out
        }
    }
}

// ---------------------------------------------------------------------------
extern "C" void kernel_launch(void* const* d_in, const int* in_sizes, int n_in,
                              void* d_out, int out_size, void* d_ws, size_t ws_size,
                              hipStream_t stream)
{
    const float* x   = (const float*)d_in[0];   // [N, C] f32
    const float* y   = (const float*)d_in[1];   // [N, C] f32
    const float* ref = (const float*)d_in[2];   // [2, H, W] f32
    const int N = in_sizes[2] / 2;              // H*W = 16384
    const int C = in_sizes[0] / N;              // 256
    float* out = (float*)d_out;                 // [2*N] warped ++ [N] confidence

    _Float16* xh = (_Float16*)d_ws;             // N*C f16  (8 MB)
    _Float16* yh = xh + (size_t)N * C;          // N*C f16  (8 MB)

    dim3 gN((unsigned)C, 2);
    normalize_cols<<<gN, 256, 0, stream>>>(x, y, xh, yh, N, C);

    const int wavesPerBlock = 8;                // 256 threads = 8 wave32
    const int numRowTiles   = N / 16;           // 1024
    const int blocks        = numRowTiles / wavesPerBlock;  // 128
    corr_softmax_warp<<<blocks, wavesPerBlock * 32, 0, stream>>>(xh, yh, ref, out, N, C);
}
